// LogicMessagePassingNetwork_11003706213179
// MI455X (gfx1250) — matrix operand order repaired and checked by reading, not compiled, not run on hardware
//
#include <hip/hip_runtime.h>

// ---------------------------------------------------------------------------
// LogicMessagePassingNetwork for MI455X (gfx1250, wave32, WMMA)
//   E=1M edges, T=2M triangles, D=64, R=50
//   out = relu(x + segsum(relu((x[ab]*x[bc]) @ Wm), ac) @ Wu),  x = feat + rel[r]
// d_out doubles as the agg scatter buffer (zeroed each call); update runs
// in-place per-row (row read to LDS before row written -> safe).
// W is staged in LDS in a fragment-contiguous tiled layout so every WMMA
// B-operand is two ds_load_b128. Scatter epilogue is branchless: padded rows
// have zero A-rows -> zero messages -> atomicAdd(+0.0f) is a no-op.
// ---------------------------------------------------------------------------

typedef __attribute__((ext_vector_type(16))) _Float16 v16h;
typedef __attribute__((ext_vector_type(8)))  _Float16 v8h;
typedef __attribute__((ext_vector_type(4)))  _Float16 v4h;
typedef __attribute__((ext_vector_type(2)))  _Float16 v2h;
typedef __attribute__((ext_vector_type(8)))  float    v8f;
typedef __attribute__((ext_vector_type(4)))  float    v4f;

#define DIM    64
#define TILE   128         // rows (triangles or edges) per block
#define NTHR   256         // 8 waves of 32
#define APITCH 72          // sA row pitch in halfs (144 B, 16B-aligned)

// ---------------------------------------------------------------- zero agg
__global__ __launch_bounds__(NTHR)
void k_zero_f4(v4f* __restrict__ p, long n4) {
    long i = (long)blockIdx.x * blockDim.x + threadIdx.x;
    long stride = (long)gridDim.x * blockDim.x;
    v4f z = {0.f, 0.f, 0.f, 0.f};
    for (; i < n4; i += stride) p[i] = z;
}

// -------------------------------------------------- W -> tiled f16 LDS fill
// sWt[(n*4 + kc)*16 + i] = W[(kc*16 + i)*64 + n];  n=0..63, kc=0..3, i=0..15
__device__ __forceinline__
void fill_sWt(_Float16* __restrict__ sWt, const float* __restrict__ W, int tid) {
    for (int idx = tid; idx < DIM * 32; idx += NTHR) {   // 2048 pairs
        int n  = idx & 63;
        int kp = idx >> 6;            // 0..31
        int k0 = kp * 2;              // even, never crosses a 16-chunk
        int kc = k0 >> 4;
        int i0 = k0 & 15;
        v2h w2;
        w2[0] = (_Float16)W[(long)k0 * DIM + n];
        w2[1] = (_Float16)W[(long)(k0 + 1) * DIM + n];
        *(v2h*)(sWt + ((n * 4 + kc) * 16 + i0)) = w2;
    }
}

// ----------------------------------------------------- shared WMMA helper
// acc[0..3] += A(16x64) @ W(64x64) for the 16 rows owned by this wave.
__device__ __forceinline__
void wave_gemm_16x64(const _Float16* __restrict__ sA,
                     const _Float16* __restrict__ sWt,
                     int t0, int g, int r, v8f acc[4]) {
    #pragma unroll
    for (int ks = 0; ks < 2; ++ks) {
        const int kb = ks * 32;
        const int kc = ks * 2 + g;                       // 16-wide K chunk
        const _Float16* arow = sA + (long)(t0 + r) * APITCH;
        v8h alo = *(const v8h*)(arow + kb + 16 * g);
        v8h ahi = *(const v8h*)(arow + kb + 16 * g + 8);
        v16h afrag = __builtin_shufflevector(alo, ahi,
            0, 1, 2, 3, 4, 5, 6, 7, 8, 9, 10, 11, 12, 13, 14, 15);
        #pragma unroll
        for (int nt = 0; nt < 4; ++nt) {
            v16h bfrag = *(const v16h*)(sWt + (((nt * 16 + r) * 4 + kc) * 16));
            acc[nt] = __builtin_amdgcn_wmma_f32_16x16x32_f16(
                false, afrag, false, bfrag, (short)0, acc[nt], false, false);
        }
    }
}

// ------------------------------------------------- triangle message kernel
__global__ __launch_bounds__(NTHR)
void k_triangle(const float* __restrict__ edge_feat,
                const float* __restrict__ rel_emb,
                const float* __restrict__ W_msg,
                const int*   __restrict__ edge_rel,
                const int*   __restrict__ ab,
                const int*   __restrict__ bc,
                const int*   __restrict__ ac,
                float*       __restrict__ agg,
                int T_) {
    __shared__ __align__(16) _Float16 sWt[DIM * DIM];
    __shared__ __align__(16) _Float16 sA[TILE][APITCH];
    __shared__ int sAB[TILE], sBC[TILE], sAC[TILE], sRA[TILE], sRB[TILE];

    const int tid = threadIdx.x;
    const int tileBase = blockIdx.x * TILE;

    fill_sWt(sWt, W_msg, tid);

    // stage triangle indices + resolved relation ids
    if (tid < TILE) {
        int t = tileBase + tid;
        if (t < T_) {
            int ea = ab[t], eb = bc[t];
            sAB[tid] = ea; sBC[tid] = eb; sAC[tid] = ac[t];
            sRA[tid] = edge_rel[ea]; sRB[tid] = edge_rel[eb];
        } else {
            sAB[tid] = 0; sBC[tid] = 0; sAC[tid] = -1; sRA[tid] = 0; sRB[tid] = 0;
        }
    }
    __syncthreads();

    // gather x[ab]*x[bc] -> f16 A-tile; float4 loads, packed b64 LDS stores.
    // Padded rows (sAC<0) are zero-filled: their WMMA output is exactly 0.
    for (int idx = tid; idx < TILE * 16; idx += NTHR) {
        int tr = idx >> 4;
        int k4 = (idx & 15) * 4;
        v4f p = {0.f, 0.f, 0.f, 0.f};
        if (sAC[tr] >= 0) {
            int ea = sAB[tr], eb = sBC[tr];
            v4f xa = *(const v4f*)(edge_feat + (long)ea * DIM + k4)
                   + *(const v4f*)(rel_emb  + (long)sRA[tr] * DIM + k4);
            v4f xb = *(const v4f*)(edge_feat + (long)eb * DIM + k4)
                   + *(const v4f*)(rel_emb  + (long)sRB[tr] * DIM + k4);
            p = xa * xb;
        }
        v4h h;
        h[0] = (_Float16)p[0]; h[1] = (_Float16)p[1];
        h[2] = (_Float16)p[2]; h[3] = (_Float16)p[3];
        *(v4h*)(&sA[tr][k4]) = h;
    }
    __syncthreads();

    // per-wave 16x64 WMMA slab
    const int wave = tid >> 5;
    const int lane = tid & 31;
    const int g    = lane >> 4;
    const int r    = lane & 15;
    const int t0   = wave * 16;

    v8f acc[4] = {};
    wave_gemm_16x64(&sA[0][0], sWt, t0, g, r, acc);

    // relu + branchless atomic scatter: one base address per output row,
    // constant byte offsets (+0/64/128/192) per N-tile. Invalid rows clamp
    // to edge 0 and add exactly +0.0f (no-op).
    #pragma unroll
    for (int j = 0; j < 8; ++j) {
        int trow = t0 + 8 * g + j;        // C/D layout: M = 8g + j
        int dst  = sAC[trow];
        dst = dst < 0 ? 0 : dst;
        float* rowp = agg + ((unsigned)(dst * DIM) + (unsigned)r);
        atomicAdd(rowp +  0, fmaxf(acc[0][j], 0.f));
        atomicAdd(rowp + 16, fmaxf(acc[1][j], 0.f));
        atomicAdd(rowp + 32, fmaxf(acc[2][j], 0.f));
        atomicAdd(rowp + 48, fmaxf(acc[3][j], 0.f));
    }
}

// ------------------------------------------------------ head-edge update
__global__ __launch_bounds__(NTHR)
void k_update(const float* __restrict__ edge_feat,
              const float* __restrict__ rel_emb,
              const float* __restrict__ W_upd,
              const int*   __restrict__ edge_rel,
              float*       __restrict__ out,   // holds agg on entry; in-place
              int E_) {
    __shared__ __align__(16) _Float16 sWt[DIM * DIM];
    __shared__ __align__(16) _Float16 sA[TILE][APITCH];

    const int tid  = threadIdx.x;
    const int base = blockIdx.x * TILE;

    fill_sWt(sWt, W_upd, tid);

    for (int idx = tid; idx < TILE * 16; idx += NTHR) {
        int row = idx >> 4;
        int k4  = (idx & 15) * 4;
        int e   = base + row;
        v4f a = {0.f, 0.f, 0.f, 0.f};
        if (e < E_) a = *(const v4f*)(out + (long)e * DIM + k4);
        v4h h;
        h[0] = (_Float16)a[0]; h[1] = (_Float16)a[1];
        h[2] = (_Float16)a[2]; h[3] = (_Float16)a[3];
        *(v4h*)(&sA[row][k4]) = h;
    }
    __syncthreads();

    const int wave = tid >> 5;
    const int lane = tid & 31;
    const int g    = lane >> 4;
    const int r    = lane & 15;
    const int t0   = wave * 16;

    v8f acc[4] = {};
    wave_gemm_16x64(&sA[0][0], sWt, t0, g, r, acc);

    // out = relu(x + agg @ Wu); one row guard + one base pointer per row,
    // immediate offsets for the 4 N-tiles.
    #pragma unroll
    for (int j = 0; j < 8; ++j) {
        int e = base + t0 + 8 * g + j;
        if (e < E_) {
            int rel = edge_rel[e];
            const float* fp = edge_feat + ((unsigned)(e * DIM) + (unsigned)r);
            const float* rp = rel_emb  + ((unsigned)(rel * DIM) + (unsigned)r);
            float*       op = out      + ((unsigned)(e * DIM) + (unsigned)r);
            op[ 0] = fmaxf(fp[ 0] + rp[ 0] + acc[0][j], 0.f);
            op[16] = fmaxf(fp[16] + rp[16] + acc[1][j], 0.f);
            op[32] = fmaxf(fp[32] + rp[32] + acc[2][j], 0.f);
            op[48] = fmaxf(fp[48] + rp[48] + acc[3][j], 0.f);
        }
    }
}

// --------------------------------------------------------------- launcher
extern "C" void kernel_launch(void* const* d_in, const int* in_sizes, int n_in,
                              void* d_out, int out_size, void* d_ws, size_t ws_size,
                              hipStream_t stream) {
    const float* edge_feat = (const float*)d_in[0];
    const float* rel_emb   = (const float*)d_in[1];
    const float* W_msg     = (const float*)d_in[2];
    const float* W_upd     = (const float*)d_in[3];
    const int*   edge_rel  = (const int*)  d_in[4];
    const int*   ab        = (const int*)  d_in[5];
    const int*   bc        = (const int*)  d_in[6];
    const int*   ac        = (const int*)  d_in[7];
    float*       out       = (float*)d_out;

    const int E_ = in_sizes[0] / DIM;
    const int T_ = in_sizes[5];

    // 1) zero agg (== d_out)
    long n4 = (long)E_ * DIM / 4;
    k_zero_f4<<<2048, NTHR, 0, stream>>>((v4f*)out, n4);

    // 2) triangle messages + atomic scatter into agg
    int tgrid = (T_ + TILE - 1) / TILE;
    k_triangle<<<tgrid, NTHR, 0, stream>>>(edge_feat, rel_emb, W_msg, edge_rel,
                                           ab, bc, ac, out, T_);

    // 3) head-edge update, in place over agg
    int ugrid = (E_ + TILE - 1) / TILE;
    k_update<<<ugrid, NTHR, 0, stream>>>(edge_feat, rel_emb, W_upd, edge_rel,
                                         out, E_);
}
